// TFDonutSwinLayer_58892591562948
// MI455X (gfx1250) — matrix operand order repaired and checked by reading, not compiled, not run on hardware
//
#include <hip/hip_runtime.h>
#include <hip/hip_bf16.h>

// ---------------------------------------------------------------------------
// Types / helpers for CDNA5 WMMA (bf16 in, f32 accumulate)
// ---------------------------------------------------------------------------
typedef __attribute__((ext_vector_type(16))) __bf16 bf16x16;
typedef __attribute__((ext_vector_type(8)))  float  f32x8;

union FragU { bf16x16 v; unsigned u[8]; };

static __device__ __forceinline__ unsigned short f2bf(float f) {
  union { float f; unsigned u; } v; v.f = f;
  unsigned r = v.u + 0x7FFFu + ((v.u >> 16) & 1u);   // round-to-nearest-even
  return (unsigned short)(r >> 16);
}

// A fragment, 16x32 bf16 (M x K). ISA layout: lane<16: M=lane, VGPRv: K=2v,2v+1
// (v<4) / K=8+2v (v>=4); lanes>=16 same M, K += 8. base points at (row0,k0),
// ld32 = row stride in dwords. All offsets dword aligned.
static __device__ __forceinline__ bf16x16 frag_a(const unsigned* base, int ld32) {
  int lane = threadIdx.x & 31;
  int r = lane & 15, half = lane >> 4;
  const unsigned* row = base + r * ld32 + half * 4;
  FragU f;
#pragma unroll
  for (int v = 0; v < 8; ++v) f.u[v] = row[(v < 4) ? v : v + 4];
  return f.v;
}

// B fragment, 32x16 bf16 (K x N), stored in LDS as [n][k] (k contiguous).
// ISA layout: lane<16: N=lane, VGPRv: K=2v,2v+1 (v<4) / K=8+2v (v>=4); lanes>=16: K += 8.
static __device__ __forceinline__ bf16x16 frag_b(const unsigned* base, int ld32) {
  int lane = threadIdx.x & 31;
  int n = lane & 15, half = lane >> 4;
  const unsigned* row = base + n * ld32 + half * 4;
  FragU f;
#pragma unroll
  for (int v = 0; v < 8; ++v) f.u[v] = row[(v < 4) ? v : v + 4];
  return f.v;
}

static __device__ __forceinline__ f32x8 wmma_bf16(bf16x16 a, bf16x16 b, f32x8 c) {
  return __builtin_amdgcn_wmma_f32_16x16x32_bf16(false, a, false, b, (short)0, c,
                                                 false, false);
}

// Async global -> LDS copy, 16 bytes, tracked by ASYNCcnt (gfx1250 TDM-lite path).
// lds_off: byte offset within workgroup LDS (low 32 bits of generic shared ptr).
static __device__ __forceinline__ void async_cp16(unsigned lds_off, const void* gaddr) {
  asm volatile("global_load_async_to_lds_b128 %0, %1, off"
               :: "v"(lds_off), "v"(gaddr) : "memory");
}

static __device__ __forceinline__ unsigned lds_addr(const void* p) {
  return (unsigned)(size_t)p;   // LDS aperture: addr[31:0] is the LDS offset
}

// ---------------------------------------------------------------------------
// Problem constants
// ---------------------------------------------------------------------------
#define Bb  8
#define Hh  128
#define Ww  128
#define Cc  384
#define NHh 12
#define HDd 32
#define WSs 8
#define SSs 4
#define Nn  64            // tokens per window
#define ROWS 131072       // B*H*W = total tokens

static __device__ __forceinline__ int band(int x) {
  return x < (Hh - WSs) ? 0 : (x < (Hh - SSs) ? 1 : 2);
}

// ---------------------------------------------------------------------------
// Weight transpose + fp32 -> bf16 convert: dst[n*K+k] = src[k*N+n]
// ---------------------------------------------------------------------------
__global__ void k_cvtT(const float* __restrict__ src, unsigned short* __restrict__ dst,
                       int K, int N) {
  long i = (long)blockIdx.x * 256 + threadIdx.x;
  if (i >= (long)K * N) return;
  int n = (int)(i / K), k = (int)(i % K);
  dst[i] = f2bf(src[(size_t)k * N + n]);
}

// biasT[h][tq][tk] = rel_bias[relIdx(tq,tk)][h]
__global__ void k_bias(const float* __restrict__ rel, float* __restrict__ biasT) {
  int i = blockIdx.x * 256 + threadIdx.x;
  if (i >= NHh * Nn * Nn) return;
  int h = i >> 12, tq = (i >> 6) & 63, tk = i & 63;
  int iq = tq >> 3, jq = tq & 7, ik = tk >> 3, jk = tk & 7;
  int idx = (iq - ik + WSs - 1) * (2 * WSs - 1) + (jq - jk + WSs - 1);
  biasT[i] = rel[idx * NHh + h];
}

// ---------------------------------------------------------------------------
// LN1 + cyclic shift + window partition -> bf16 xw (row = window-token space)
// ---------------------------------------------------------------------------
__global__ __launch_bounds__(128) void k_ln1win(
    const float* __restrict__ hid, const float* __restrict__ g1,
    const float* __restrict__ b1, unsigned short* __restrict__ out) {
  __shared__ float red[128], red2[128];
  int r = blockIdx.x, t = threadIdx.x;
  int b = r >> 14;
  int within = r & 16383;
  int wIdx = within >> 6, tok = within & 63;
  int wh = wIdx >> 4, ww = wIdx & 15;
  int hsh = wh * 8 + (tok >> 3), wsh = ww * 8 + (tok & 7);
  int h0 = (hsh + SSs) & 127, w0 = (wsh + SSs) & 127;    // roll(-SS)
  const float* src = hid + ((size_t)b * 16384 + h0 * 128 + w0) * Cc;
  float x[3]; float s = 0.f, ss = 0.f;
#pragma unroll
  for (int j = 0; j < 3; ++j) { x[j] = src[t + 128 * j]; s += x[j]; ss += x[j] * x[j]; }
  red[t] = s; red2[t] = ss;
  __syncthreads();
  for (int st = 64; st > 0; st >>= 1) {
    if (t < st) { red[t] += red[t + st]; red2[t] += red2[t + st]; }
    __syncthreads();
  }
  float mu = red[0] * (1.f / Cc);
  float var = red2[0] * (1.f / Cc) - mu * mu;
  float rs = rsqrtf(var + 1e-5f);
  unsigned short* dst = out + (size_t)r * Cc;
#pragma unroll
  for (int j = 0; j < 3; ++j) {
    int e = t + 128 * j;
    dst[e] = f2bf((x[j] - mu) * rs * g1[e] + b1[e]);
  }
}

// LN2 on fp32 x (in d_out) -> bf16
__global__ __launch_bounds__(128) void k_ln2(
    const float* __restrict__ xin, const float* __restrict__ g2,
    const float* __restrict__ b2, unsigned short* __restrict__ out) {
  __shared__ float red[128], red2[128];
  int r = blockIdx.x, t = threadIdx.x;
  const float* src = xin + (size_t)r * Cc;
  float x[3]; float s = 0.f, ss = 0.f;
#pragma unroll
  for (int j = 0; j < 3; ++j) { x[j] = src[t + 128 * j]; s += x[j]; ss += x[j] * x[j]; }
  red[t] = s; red2[t] = ss;
  __syncthreads();
  for (int st = 64; st > 0; st >>= 1) {
    if (t < st) { red[t] += red[t + st]; red2[t] += red2[t + st]; }
    __syncthreads();
  }
  float mu = red[0] * (1.f / Cc);
  float var = red2[0] * (1.f / Cc) - mu * mu;
  float rs = rsqrtf(var + 1e-5f);
  unsigned short* dst = out + (size_t)r * Cc;
#pragma unroll
  for (int j = 0; j < 3; ++j) {
    int e = t + 128 * j;
    dst[e] = f2bf((x[j] - mu) * rs * g2[e] + b2[e]);
  }
}

// ---------------------------------------------------------------------------
// Tiled bf16 WMMA GEMM with double-buffered LDS + async global->LDS copies.
// out = epilogue(A[MxK] @ Bt[NxK]^T + bias)
// Block tile 128x128, 8 waves (4x2), wave tile 32x64, K-step 32.
// MODE 0: bf16 store   MODE 1: GELU + bf16 store
// MODE 2: window-reverse + unshift + shortcut residual, fp32 store
// MODE 3: fp32 accumulate into outp (reads previous value)
// ---------------------------------------------------------------------------
template <int MODE>
__global__ __launch_bounds__(256) void k_gemm(
    const unsigned short* __restrict__ A, const unsigned short* __restrict__ Bt,
    const float* __restrict__ bias, void* __restrict__ outp,
    const float* __restrict__ aux, int M, int N, int K) {
  __shared__ unsigned short As[2][128 * 32];
  __shared__ unsigned short Bs[2][128 * 32];
  int tid = threadIdx.x;
  int lane = tid & 31, wave = tid >> 5;
  int wm = (wave >> 1) * 32, wn = (wave & 1) * 64;
  int r = lane & 15, half = lane >> 4;
  size_t blockM = (size_t)blockIdx.y * 128;
  size_t blockN = (size_t)blockIdx.x * 128;
  f32x8 acc[2][4];
#pragma unroll
  for (int mi = 0; mi < 2; ++mi)
#pragma unroll
    for (int ni = 0; ni < 4; ++ni) acc[mi][ni] = (f32x8){0, 0, 0, 0, 0, 0, 0, 0};

  const uint4* A4 = (const uint4*)A;
  const uint4* B4 = (const uint4*)Bt;

  // per-thread async copy of one 128x32 bf16 A tile + B tile (4 x b128 per thread)
  auto issue_tile = [&](int k0, int buf) {
    unsigned asb = lds_addr(&As[buf][0]);
    unsigned bsb = lds_addr(&Bs[buf][0]);
#pragma unroll
    for (int c = tid; c < 512; c += 256) {
      int row = c >> 2, seg = c & 3;
      const uint4* ga = A4 + (((blockM + row) * (size_t)K + k0) >> 3) + seg;
      const uint4* gb = B4 + (((blockN + row) * (size_t)K + k0) >> 3) + seg;
      async_cp16(asb + (unsigned)c * 16u, ga);
      async_cp16(bsb + (unsigned)c * 16u, gb);
    }
  };

  int nIter = K >> 5;
  issue_tile(0, 0);
  for (int it = 0; it < nIter; ++it) {
    int buf = it & 1;
    if (it + 1 < nIter) {
      issue_tile((it + 1) << 5, buf ^ 1);
      asm volatile("s_wait_asynccnt 4" ::: "memory");   // current tile (4 ops) done
    } else {
      asm volatile("s_wait_asynccnt 0" ::: "memory");
    }
    __syncthreads();                                    // all waves' copies visible

    const unsigned* As32 = (const unsigned*)&As[buf][0];
    const unsigned* Bs32 = (const unsigned*)&Bs[buf][0];
    bf16x16 af[2];
#pragma unroll
    for (int mi = 0; mi < 2; ++mi) af[mi] = frag_a(As32 + (wm + mi * 16) * 16, 16);
#pragma unroll
    for (int ni = 0; ni < 4; ++ni) {
      bf16x16 bfr = frag_b(Bs32 + (wn + ni * 16) * 16, 16);
#pragma unroll
      for (int mi = 0; mi < 2; ++mi) acc[mi][ni] = wmma_bf16(af[mi], bfr, acc[mi][ni]);
    }
    __syncthreads();   // reads done before this buffer is overwritten (it+2)
  }

#pragma unroll
  for (int mi = 0; mi < 2; ++mi) {
#pragma unroll
    for (int ni = 0; ni < 4; ++ni) {
      size_t col = blockN + wn + ni * 16 + r;
      float bv = bias[col];
#pragma unroll
      for (int g = 0; g < 8; ++g) {
        size_t row = blockM + wm + mi * 16 + half * 8 + g;
        float v = acc[mi][ni][g] + bv;
        if (MODE == 0) {
          ((unsigned short*)outp)[row * N + col] = f2bf(v);
        } else if (MODE == 1) {
          float gl = 0.5f * v * (1.0f + erff(v * 0.70710678118654752f));
          ((unsigned short*)outp)[row * N + col] = f2bf(gl);
        } else if (MODE == 2) {
          int b = (int)(row >> 14);
          int within = (int)(row & 16383);
          int wIdx = within >> 6, t = within & 63;
          int wh = wIdx >> 4, ww = wIdx & 15;
          int hsh = wh * 8 + (t >> 3), wsh = ww * 8 + (t & 7);
          int ho = (hsh + SSs) & 127, wo_ = (wsh + SSs) & 127;  // roll(+SS)
          size_t prow = (size_t)b * 16384 + ho * 128 + wo_;
          float* o = (float*)outp;
          o[prow * Cc + col] = aux[prow * Cc + col] + v;        // + shortcut
        } else {
          float* o = (float*)outp;
          size_t idx = row * N + col;
          o[idx] = o[idx] + v;                                  // x + mlp
        }
      }
    }
  }
}

// ---------------------------------------------------------------------------
// Windowed attention, one wave per (window, head).
// S = Q Kt / sqrt(32) + rel_bias + shift_mask ; softmax ; ctx = P V
// ---------------------------------------------------------------------------
__global__ __launch_bounds__(32) void k_attn(
    const unsigned short* __restrict__ Q, const unsigned short* __restrict__ Km,
    const unsigned short* __restrict__ V, const float* __restrict__ biasT,
    unsigned short* __restrict__ ctx) {
  __shared__ unsigned short Qs[Nn * HDd];   // 64x32 (M x K) row-major
  __shared__ unsigned short Ks[Nn * HDd];   // 64x32 == B as [n][k] for S
  __shared__ unsigned short Vts[HDd * Nn];  // 32x64 == B as [n][k] for PV
  __shared__ float          Ss[Nn * Nn];    // scores f32
  __shared__ unsigned short Ps[Nn * Nn];    // probs bf16

  int win = blockIdx.x, head = blockIdx.y;
  int lane = threadIdx.x;
  size_t base = (size_t)win * Nn * Cc + head * HDd;

  // stage Q,K rows (2 rows / lane, 4 x 16B each)
#pragma unroll
  for (int rr = 0; rr < 2; ++rr) {
    int t = lane * 2 + rr;
    const uint4* gq = (const uint4*)(Q + base + (size_t)t * Cc);
    const uint4* gk = (const uint4*)(Km + base + (size_t)t * Cc);
    uint4* sq = (uint4*)(Qs + t * HDd);
    uint4* sk = (uint4*)(Ks + t * HDd);
#pragma unroll
    for (int c = 0; c < 4; ++c) { sq[c] = gq[c]; sk[c] = gk[c]; }
  }
  // stage V transposed: Vts[d][t]
  {
    int d = lane;
    for (int t = 0; t < Nn; ++t)
      Vts[d * Nn + t] = V[base + (size_t)t * Cc + d];
  }
  __syncthreads();

  int r = lane & 15, half = lane >> 4;
  const unsigned* Qs32 = (const unsigned*)Qs;
  const unsigned* Ks32 = (const unsigned*)Ks;
  int wIdx = win & 255, wh = wIdx >> 4, ww = wIdx & 15;

  // S = Q @ K^T  (M=64,N=64,K=32), epilogue: scale + bias + mask -> LDS
#pragma unroll
  for (int mi = 0; mi < 4; ++mi) {
    bf16x16 af = frag_a(Qs32 + (mi * 16) * 16, 16);
    // per-row (tq) region ids, computed once per mi
    int rqv[8];
#pragma unroll
    for (int g = 0; g < 8; ++g) {
      int tq = mi * 16 + half * 8 + g;
      int hq = wh * 8 + (tq >> 3), wq_ = ww * 8 + (tq & 7);
      rqv[g] = band(hq) * 3 + band(wq_);
    }
#pragma unroll
    for (int ni = 0; ni < 4; ++ni) {
      bf16x16 bfr = frag_b(Ks32 + (ni * 16) * 16, 16);
      f32x8 z = (f32x8){0, 0, 0, 0, 0, 0, 0, 0};
      f32x8 acc = wmma_bf16(af, bfr, z);
      int tk = ni * 16 + r;
      int hk = wh * 8 + (tk >> 3), wk_ = ww * 8 + (tk & 7);
      int rk = band(hk) * 3 + band(wk_);
#pragma unroll
      for (int g = 0; g < 8; ++g) {
        int tq = mi * 16 + half * 8 + g;
        float v = acc[g] * 0.17677669529663687f
                + biasT[head * (Nn * Nn) + tq * Nn + tk]
                + (rqv[g] != rk ? -100.0f : 0.0f);
        Ss[tq * Nn + tk] = v;
      }
    }
  }
  __syncthreads();

  // row softmax (2 rows per lane), write bf16 probs
  for (int row = lane; row < Nn; row += 32) {
    float m = -1e30f;
    for (int c = 0; c < Nn; ++c) m = fmaxf(m, Ss[row * Nn + c]);
    float s = 0.f;
    for (int c = 0; c < Nn; ++c) {
      float e = __expf(Ss[row * Nn + c] - m);
      Ss[row * Nn + c] = e;
      s += e;
    }
    float inv = 1.f / s;
    for (int c = 0; c < Nn; ++c) Ps[row * Nn + c] = f2bf(Ss[row * Nn + c] * inv);
  }
  __syncthreads();

  // ctx = P @ V  (M=64,N=32,K=64): 2 chained WMMAs per 16x16 tile
  const unsigned* Ps32 = (const unsigned*)Ps;
  const unsigned* Vt32 = (const unsigned*)Vts;
#pragma unroll
  for (int mi = 0; mi < 4; ++mi) {
    bf16x16 a0 = frag_a(Ps32 + mi * 16 * 32, 32);        // k0 = 0
    bf16x16 a1 = frag_a(Ps32 + mi * 16 * 32 + 16, 32);   // k0 = 32
#pragma unroll
    for (int ni = 0; ni < 2; ++ni) {
      bf16x16 b0 = frag_b(Vt32 + ni * 16 * 32, 32);
      bf16x16 b1 = frag_b(Vt32 + ni * 16 * 32 + 16, 32);
      f32x8 z = (f32x8){0, 0, 0, 0, 0, 0, 0, 0};
      f32x8 acc = wmma_bf16(a0, b0, z);
      acc = wmma_bf16(a1, b1, acc);
#pragma unroll
      for (int g = 0; g < 8; ++g) {
        int row = mi * 16 + half * 8 + g;
        int col = ni * 16 + r;
        ctx[base + (size_t)row * Cc + col] = f2bf(acc[g]);
      }
    }
  }
}

// ---------------------------------------------------------------------------
// Host-side orchestration
// ---------------------------------------------------------------------------
extern "C" void kernel_launch(void* const* d_in, const int* in_sizes, int n_in,
                              void* d_out, int out_size, void* d_ws, size_t ws_size,
                              hipStream_t stream) {
  const float* hid  = (const float*)d_in[0];
  const float* ln1g = (const float*)d_in[1];
  const float* ln1b = (const float*)d_in[2];
  const float* wq   = (const float*)d_in[3];
  const float* bq   = (const float*)d_in[4];
  const float* wk   = (const float*)d_in[5];
  const float* bk   = (const float*)d_in[6];
  const float* wv   = (const float*)d_in[7];
  const float* bv   = (const float*)d_in[8];
  const float* rel  = (const float*)d_in[9];
  const float* wo   = (const float*)d_in[10];
  const float* bo   = (const float*)d_in[11];
  const float* ln2g = (const float*)d_in[12];
  const float* ln2b = (const float*)d_in[13];
  const float* w1   = (const float*)d_in[14];
  const float* b1   = (const float*)d_in[15];
  const float* w2   = (const float*)d_in[16];
  const float* b2   = (const float*)d_in[17];
  float* out = (float*)d_out;

  char* ws = (char*)d_ws;
  const size_t SZ = (size_t)ROWS * Cc * 2;            // one bf16 activation buffer
  unsigned short* xw = (unsigned short*)ws;           // xw, later LN2 output
  unsigned short* Qb = (unsigned short*)(ws + 1 * SZ);
  unsigned short* Kb = (unsigned short*)(ws + 2 * SZ);
  unsigned short* Vb = (unsigned short*)(ws + 3 * SZ);
  unsigned short* Cx = (unsigned short*)(ws + 4 * SZ);
  unsigned short* h1 = Qb;                            // reuse Q..Cx: exactly ROWS*1536 bf16
  size_t o = 5 * SZ;
  unsigned short* wqT = (unsigned short*)(ws + o); o += (size_t)Cc * Cc * 2;
  unsigned short* wkT = (unsigned short*)(ws + o); o += (size_t)Cc * Cc * 2;
  unsigned short* wvT = (unsigned short*)(ws + o); o += (size_t)Cc * Cc * 2;
  unsigned short* woT = (unsigned short*)(ws + o); o += (size_t)Cc * Cc * 2;
  unsigned short* w1T = (unsigned short*)(ws + o); o += (size_t)Cc * 4 * Cc * 2;
  unsigned short* w2T = (unsigned short*)(ws + o); o += (size_t)Cc * 4 * Cc * 2;
  float* biasT = (float*)(ws + o);

  // 1) weight convert/transpose + bias table
  k_cvtT<<<(Cc * Cc + 255) / 256, 256, 0, stream>>>(wq, wqT, Cc, Cc);
  k_cvtT<<<(Cc * Cc + 255) / 256, 256, 0, stream>>>(wk, wkT, Cc, Cc);
  k_cvtT<<<(Cc * Cc + 255) / 256, 256, 0, stream>>>(wv, wvT, Cc, Cc);
  k_cvtT<<<(Cc * Cc + 255) / 256, 256, 0, stream>>>(wo, woT, Cc, Cc);
  k_cvtT<<<(Cc * 4 * Cc + 255) / 256, 256, 0, stream>>>(w1, w1T, Cc, 4 * Cc);
  k_cvtT<<<(Cc * 4 * Cc + 255) / 256, 256, 0, stream>>>(w2, w2T, 4 * Cc, Cc);
  k_bias<<<(NHh * Nn * Nn + 255) / 256, 256, 0, stream>>>(rel, biasT);

  // 2) LN1 + shift + window partition
  k_ln1win<<<ROWS, 128, 0, stream>>>(hid, ln1g, ln1b, xw);

  // 3) QKV projections
  dim3 g384(Cc / 128, ROWS / 128);
  dim3 g1536(4 * Cc / 128, ROWS / 128);
  k_gemm<0><<<g384, 256, 0, stream>>>(xw, wqT, bq, Qb, nullptr, ROWS, Cc, Cc);
  k_gemm<0><<<g384, 256, 0, stream>>>(xw, wkT, bk, Kb, nullptr, ROWS, Cc, Cc);
  k_gemm<0><<<g384, 256, 0, stream>>>(xw, wvT, bv, Vb, nullptr, ROWS, Cc, Cc);

  // 4) windowed attention
  k_attn<<<dim3(Bb * 256, NHh), 32, 0, stream>>>(Qb, Kb, Vb, biasT, Cx);

  // 5) output proj + window reverse + unshift + residual -> x in d_out
  k_gemm<2><<<g384, 256, 0, stream>>>(Cx, woT, bo, out, hid, ROWS, Cc, Cc);

  // 6) LN2 -> bf16, MLP
  k_ln2<<<ROWS, 128, 0, stream>>>(out, ln2g, ln2b, xw);
  k_gemm<1><<<g1536, 256, 0, stream>>>(xw, w1T, b1, h1, nullptr, ROWS, 4 * Cc, Cc);
  k_gemm<3><<<g384, 256, 0, stream>>>(h1, w2T, b2, out, nullptr, ROWS, Cc, 4 * Cc);
}
